// FullAttention_43181601194900
// MI455X (gfx1250) — matrix operand order repaired and checked by reading, not compiled, not run on hardware
//
#include <hip/hip_runtime.h>

// ---------------------------------------------------------------------------
// Flash attention for gfx1250 (MI455X), bf16 WMMA, fp32 accumulate.
// B=4, L=S=2048, H=8, E=D=64.  out[b,l,h,d]
// Softmax runs in the log2 domain (scale*log2e folded into Q), row reductions
// use DPP16 butterflies (VALU, co-executes with WMMA) instead of ds_bpermute.
// ---------------------------------------------------------------------------

typedef __attribute__((ext_vector_type(16))) __bf16 v16bf;
typedef __attribute__((ext_vector_type(8)))  __bf16 v8bf;
typedef __attribute__((ext_vector_type(4)))  __bf16 v4bf;
typedef __attribute__((ext_vector_type(2)))  __bf16 v2bf;
typedef __attribute__((ext_vector_type(8)))  float  v8f;

namespace {
constexpr int kB = 4, kL = 2048, kS = 2048, kH = 8, kE = 64, kD = 64;
constexpr int kWaves = 8;                                // 256 threads / wave32
constexpr int kQRowsPerWave = 16;                        // WMMA M tile
constexpr int kQRowsPerBlock = kWaves * kQRowsPerWave;   // 128
constexpr int kSTile = 64;       // S columns staged per outer iteration
constexpr int kLdsStride = 80;   // bf16/row for K,VT (64+16 pad; 160B rows, 32B-aligned frags)
constexpr int kPStride = 72;     // bf16/row for P scratch (64+8 pad; 144B rows, 16B-aligned chunks)
}

// ---- DPP16 butterfly allreduce within each 16-lane row --------------------
template <int CTRL>
__device__ __forceinline__ float dpp_xchg(float x) {
    const int v = __builtin_amdgcn_update_dpp(
        __float_as_int(x), __float_as_int(x), CTRL, 0xf, 0xf, true);
    return __int_as_float(v);
}
__device__ __forceinline__ float rowmax16(float v) {
    v = fmaxf(v, dpp_xchg<0xB1>(v));    // quad_perm(1,0,3,2): xor 1
    v = fmaxf(v, dpp_xchg<0x4E>(v));    // quad_perm(2,3,0,1): xor 2
    v = fmaxf(v, dpp_xchg<0x141>(v));   // row_half_mirror:    xor 7 (merges quads)
    v = fmaxf(v, dpp_xchg<0x140>(v));   // row_mirror:         xor 15 (merges halves)
    return v;
}
__device__ __forceinline__ float rowsum16(float v) {
    v += dpp_xchg<0xB1>(v);
    v += dpp_xchg<0x4E>(v);
    v += dpp_xchg<0x141>(v);
    v += dpp_xchg<0x140>(v);
    return v;
}

__global__ __launch_bounds__(256)
void fa_fwd_bf16wmma(const float* __restrict__ Q,
                     const float* __restrict__ K,
                     const float* __restrict__ V,
                     float* __restrict__ O)
{
    __shared__ __bf16 sK [kSTile * kLdsStride];              // [s_local][e]      10.0 KB
    __shared__ __bf16 sVT[kD    * kLdsStride];               // [d][s_local] (T)  10.0 KB
    __shared__ __bf16 sP [kWaves][kQRowsPerWave * kPStride]; // per-wave P 16x64  18.0 KB

    const int t    = threadIdx.x;
    const int wave = t >> 5;
    const int lane = t & 31;
    const int half = lane >> 4;     // 0: lanes 0-15, 1: lanes 16-31
    const int ln   = lane & 15;

    int blk = blockIdx.x;                       // b*H*(L/128) blocks
    const int qblock = blk % (kL / kQRowsPerBlock); blk /= (kL / kQRowsPerBlock);
    const int h      = blk % kH;                    blk /= kH;
    const int b      = blk;
    const int qbase  = qblock * kQRowsPerBlock + wave * kQRowsPerWave;

    // scale * log2(e): whole softmax runs in the exp2 domain, folded into Q.
    const float qscale = 0.125f * 1.44269504088896340736f;

    // ---- Q A-fragments (16x32 bf16, contiguous-k A operand), 2 fragments cover E=64.
    // A elem e -> (m = lane%16, k = 16*half + e)  [contiguous 16]
    v16bf qa[2];
    {
        const float* qrow = Q + (((size_t)b * kL + (qbase + ln)) * kH + h) * kE;
        #pragma unroll
        for (int kc = 0; kc < 2; ++kc) {
            #pragma unroll
            for (int e = 0; e < 16; ++e) {
                const int k = kc * 32 + 16 * half + e;
                qa[kc][e] = (__bf16)(qrow[k] * qscale);
            }
        }
    }

    // Softmax running state: lane holds 8 rows (r -> m = r + 8*half in C-layout).
    float mrun[8], lrun[8];
    v8f   acc[4];                                // O tile 16x64 = 4 C-frags
    #pragma unroll
    for (int r = 0; r < 8; ++r) { mrun[r] = -1e30f; lrun[r] = 0.0f; }
    #pragma unroll
    for (int d = 0; d < 4; ++d)
        #pragma unroll
        for (int r = 0; r < 8; ++r) acc[d][r] = 0.0f;

    const float* kbase = K + (((size_t)b * kS) * kH + h) * kE;   // + s*kH*kE
    const float* vbase = V + (((size_t)b * kS) * kH + h) * kD;   // + s*kH*kD

    for (int s0 = 0; s0 < kS; s0 += kSTile) {
        __syncthreads();
        // ---- cooperative staging: 64x64 K (row-major) + 64x64 V (transposed),
        //      two consecutive s-rows per task so VT stores pack as b32 pairs.
        #pragma unroll
        for (int i = 0; i < 2; ++i) {
            const int fidx = t + i * 256;        // 0..511: 32 row-pairs x 16 float4
            const int sr   = (fidx >> 4) << 1;   // even row
            const int c4   = (fidx & 15) << 2;
            const float* kr0 = kbase + (size_t)(s0 + sr) * (kH * kE) + c4;
            const float* vr0 = vbase + (size_t)(s0 + sr) * (kH * kD) + c4;

            const float4 fk0 = *(const float4*)(kr0);
            const float4 fk1 = *(const float4*)(kr0 + kH * kE);
            v4bf pk0, pk1;
            pk0[0] = (__bf16)fk0.x; pk0[1] = (__bf16)fk0.y;
            pk0[2] = (__bf16)fk0.z; pk0[3] = (__bf16)fk0.w;
            pk1[0] = (__bf16)fk1.x; pk1[1] = (__bf16)fk1.y;
            pk1[2] = (__bf16)fk1.z; pk1[3] = (__bf16)fk1.w;
            *(v4bf*)(&sK[(sr    ) * kLdsStride + c4]) = pk0;
            *(v4bf*)(&sK[(sr + 1) * kLdsStride + c4]) = pk1;

            const float4 fv0 = *(const float4*)(vr0);
            const float4 fv1 = *(const float4*)(vr0 + kH * kD);
            const float a0[4] = {fv0.x, fv0.y, fv0.z, fv0.w};
            const float a1[4] = {fv1.x, fv1.y, fv1.z, fv1.w};
            #pragma unroll
            for (int j = 0; j < 4; ++j) {
                v2bf pr; pr[0] = (__bf16)a0[j]; pr[1] = (__bf16)a1[j];
                *(v2bf*)(&sVT[(c4 + j) * kLdsStride + sr]) = pr;  // sr even: 4B aligned
            }

            // L2 prefetch of the next chunk (hidden behind the WMMA-bound loop)
            if (s0 + kSTile < kS) {
                __builtin_prefetch(kr0 + (size_t)kSTile * (kH * kE), 0, 1);
                __builtin_prefetch(vr0 + (size_t)kSTile * (kH * kD), 0, 1);
            }
        }
        __syncthreads();

        // ---- scores for all 64 staged columns: 4 tiles of 16, S = Q x K^T
        v8f sfrag[4];
        #pragma unroll
        for (int j = 0; j < 4; ++j) {
            v8f c;
            #pragma unroll
            for (int r = 0; r < 8; ++r) c[r] = 0.0f;
            #pragma unroll
            for (int kc = 0; kc < 2; ++kc) {
                // B elem e -> (k = e + 16*half within chunk, n = lane%16): 16 contig bf16
                const v16bf kb = *(const v16bf*)(
                    &sK[(j * 16 + ln) * kLdsStride + kc * 32 + 16 * half]);
                c = __builtin_amdgcn_wmma_f32_16x16x32_bf16(
                        false, qa[kc], false, kb, (short)0, c, false, false);
            }
            sfrag[j] = c;
        }

        // ---- one online-softmax pass over the 64 columns (log2 domain)
        float alpha[8];
        #pragma unroll
        for (int r = 0; r < 8; ++r) {
            float v = fmaxf(fmaxf(sfrag[0][r], sfrag[1][r]),
                            fmaxf(sfrag[2][r], sfrag[3][r]));
            v = rowmax16(v);
            const float mn = fmaxf(mrun[r], v);
            alpha[r] = exp2f(mrun[r] - mn);
            mrun[r]  = mn;
        }
        #pragma unroll
        for (int r = 0; r < 8; ++r) {
            const float p0 = exp2f(sfrag[0][r] - mrun[r]);
            const float p1 = exp2f(sfrag[1][r] - mrun[r]);
            const float p2 = exp2f(sfrag[2][r] - mrun[r]);
            const float p3 = exp2f(sfrag[3][r] - mrun[r]);
            sfrag[0][r] = p0; sfrag[1][r] = p1;
            sfrag[2][r] = p2; sfrag[3][r] = p3;
            const float rs = rowsum16((p0 + p1) + (p2 + p3));
            lrun[r] = lrun[r] * alpha[r] + rs;
        }
        #pragma unroll
        for (int d = 0; d < 4; ++d)
            #pragma unroll
            for (int r = 0; r < 8; ++r) acc[d][r] *= alpha[r];

        // ---- C-layout -> A-layout for P (16x64) via per-wave LDS scratch
        __bf16* pw = &sP[wave][0];
        #pragma unroll
        for (int r = 0; r < 8; ++r) {
            const int m = r + 8 * half;
            #pragma unroll
            for (int j = 0; j < 4; ++j)
                pw[m * kPStride + j * 16 + ln] = (__bf16)sfrag[j][r];
        }
        // intra-wave LDS store->load hazard: drain DS counter explicitly
        asm volatile("s_wait_dscnt 0" ::: "memory");

        v16bf pa[2];
        #pragma unroll
        for (int kc = 0; kc < 2; ++kc) {
            const v8bf lo = *(const v8bf*)(&pw[ln * kPStride + kc * 32 + 16 * half]);
            const v8bf hi = *(const v8bf*)(&pw[ln * kPStride + kc * 32 + 16 * half + 8]);
            #pragma unroll
            for (int e = 0; e < 8; ++e) { pa[kc][e] = lo[e]; pa[kc][e + 8] = hi[e]; }
        }

        // ---- O += P(16x64) x V(64x64): 4 D-tiles x 2 K-chunks
        #pragma unroll
        for (int d = 0; d < 4; ++d) {
            #pragma unroll
            for (int kc = 0; kc < 2; ++kc) {
                const v16bf vb = *(const v16bf*)(
                    &sVT[(d * 16 + ln) * kLdsStride + kc * 32 + 16 * half]);
                acc[d] = __builtin_amdgcn_wmma_f32_16x16x32_bf16(
                             false, pa[kc], false, vb, (short)0, acc[d], false, false);
            }
        }
    }

    // ---- epilogue: normalize (one rcp per row) and store O[b, l, h, d] fp32
    float inv[8];
    #pragma unroll
    for (int r = 0; r < 8; ++r) inv[r] = 1.0f / lrun[r];
    #pragma unroll
    for (int d = 0; d < 4; ++d) {
        #pragma unroll
        for (int r = 0; r < 8; ++r) {
            const int row = qbase + r + 8 * half;
            O[(((size_t)b * kL + row) * kH + h) * kD + d * 16 + ln] = acc[d][r] * inv[r];
        }
    }
}

extern "C" void kernel_launch(void* const* d_in, const int* in_sizes, int n_in,
                              void* d_out, int out_size, void* d_ws, size_t ws_size,
                              hipStream_t stream) {
    (void)in_sizes; (void)n_in; (void)out_size; (void)d_ws; (void)ws_size;
    const float* Q = (const float*)d_in[0];
    const float* K = (const float*)d_in[1];
    const float* V = (const float*)d_in[2];
    float* O = (float*)d_out;

    const int grid = kB * kH * (kL / kQRowsPerBlock);   // 4*8*16 = 512 blocks
    fa_fwd_bf16wmma<<<grid, 256, 0, stream>>>(Q, K, V, O);
}
